// DAGG_25890062860597
// MI455X (gfx1250) — matrix operand order repaired and checked by reading, not compiled, not run on hardware
//
#include <hip/hip_runtime.h>
#include <math.h>

typedef __attribute__((ext_vector_type(16))) _Float16 v16h;
typedef __attribute__((ext_vector_type(8)))  float    v8f;

#define Sb   32
#define Nn   128
#define Nm1  127
#define Dd   256
#define Ff   516
#define FPAD 544
#define TRIE 8128
#define Mrows 4064   // Sb * Nm1

// ---- optional CDNA5 async global->LDS staging (guarded; fallback = sync copy) ----
#if defined(__has_builtin)
#if __has_builtin(__builtin_amdgcn_global_load_async_to_lds_b128) && \
    __has_builtin(__builtin_amdgcn_s_wait_asynccnt)
#define USE_ASYNC_LDS 1
#endif
#endif
#ifndef USE_ASYNC_LDS
#define USE_ASYNC_LDS 0
#endif

#if USE_ASYNC_LDS
// ROCm clang declares this builtin with generic pointers to 16B int vectors
// (diagnostic: param type 'int __vector(4) *').
typedef int v4i_vs __attribute__((vector_size(16)));
__device__ inline void async_copy_b128(const void* g, void* l) {
  __builtin_amdgcn_global_load_async_to_lds_b128((v4i_vs*)g, (v4i_vs*)l, 0, 0);
}
#endif

// ---------------- WMMA fragment loaders (CDNA5 16x16x32 f16) ----------------

// A fragment: 16x32 f16 tile from row-major matrix, ld in halves.
// lane L holds row (L&15); K = {0..7,16..23} for L<16, {8..15,24..31} for L>=16.
__device__ inline v16h frag_a(const _Float16* base, int ld, int lane) {
  const _Float16* q = base + (lane & 15) * ld + ((lane >> 4) << 3);
  v16h a;
#pragma unroll
  for (int i = 0; i < 8; ++i) a[i] = q[i];
#pragma unroll
  for (int i = 0; i < 8; ++i) a[8 + i] = q[16 + i];
  return a;
}

// B fragment: 32x16 f16 (K x N) sourced from row-major W (N x K) -> column-major B.
// lane L holds column (L&15); K = {0..7,16..23} for L<16, {8..15,24..31} for L>=16.
__device__ inline v16h frag_b(const _Float16* base, int ld, int lane) {
  const _Float16* q = base + (lane & 15) * ld + ((lane >> 4) << 3);
  v16h b;
#pragma unroll
  for (int i = 0; i < 8; ++i) b[i] = q[i];
#pragma unroll
  for (int i = 0; i < 8; ++i) b[8 + i] = q[16 + i];
  return b;
}

// ---------------- small kernels ----------------

__global__ void k_zero(float* out) {
  if (threadIdx.x < Sb * 2) out[threadIdx.x] = 0.f;
}

// convert f32 -> f16, row-major, with optional zero padding of extra columns
__global__ void k_cvt16(const float* __restrict__ src, int sld,
                        _Float16* __restrict__ dst, int dld, int rows, int ccopy) {
  int total = rows * dld;
  for (int idx = blockIdx.x * blockDim.x + threadIdx.x; idx < total;
       idx += gridDim.x * blockDim.x) {
    int r = idx / dld, c = idx % dld;
    dst[idx] = (c < ccopy) ? (_Float16)src[(size_t)r * sld + c] : (_Float16)0.f;
  }
}

// x[:,1:,:] (f32, (32,128,516)) -> padded f16 (4064 x 544)
__global__ void k_cvt_x(const float* __restrict__ x, _Float16* __restrict__ xp) {
  int total = Mrows * FPAD;
  for (int idx = blockIdx.x * blockDim.x + threadIdx.x; idx < total;
       idx += gridDim.x * blockDim.x) {
    int m = idx / FPAD, c = idx % FPAD;
    int s = m / Nm1, n = m % Nm1 + 1;
    xp[idx] = (c < Ff) ? (_Float16)x[((size_t)s * Nn + n) * Ff + c] : (_Float16)0.f;
  }
}

// tiny compact-content MLP -> c0; broadcast c0 (f16) into decoder input slot t=0
__global__ void k_c0(const float* z, const float* ini,
                     const float* W0, const float* b0,
                     const float* W1, const float* b1,
                     const float* W2, const float* b2,
                     float* c0f, _Float16* content16) {
  __shared__ float in[512];
  __shared__ float t1[256];
  __shared__ float t2[128];
  int tid = threadIdx.x;  // 256 threads
  in[tid] = z[tid];
  in[256 + tid] = ini[tid];
  __syncthreads();
  {
    float a = b0[tid];
    for (int k = 0; k < 512; ++k) a += W0[tid * 512 + k] * in[k];
    t1[tid] = fmaxf(a, 0.f);
  }
  __syncthreads();
  if (tid < 128) {
    float a = b1[tid];
    for (int k = 0; k < 256; ++k) a += W1[tid * 256 + k] * t1[k];
    t2[tid] = fmaxf(a, 0.f);
  }
  __syncthreads();
  {
    float a = b2[tid];
    for (int k = 0; k < 128; ++k) a += W2[tid * 128 + k] * t2[k];
    c0f[tid] = a;
    _Float16 h = (_Float16)a;
    for (int s = 0; s < Sb; ++s)
      content16[(size_t)s * Nm1 * Dd + tid] = h;  // row t=0 per sample
  }
}

// ---------------- generic f16 WMMA GEMM: out = act(A @ B^T + bias) ----------------
// A: (M x KH) f16 row-major, lda == KH. B: (N x KH) f16 row-major (torch Linear weight).
// Block: 128 threads (4 waves); block tile 16 rows x 256 cols (wave = 64 cols).
template <int KH, bool RELU, bool BIAS, bool OUTF16>
__global__ __launch_bounds__(128) void k_gemm16(const _Float16* __restrict__ A,
                                                const _Float16* __restrict__ B,
                                                const float* __restrict__ bias,
                                                _Float16* __restrict__ outh,
                                                float* __restrict__ outf, int ldc) {
  __shared__ _Float16 As[16 * KH];
  const int tid = threadIdx.x;
  const int lane = tid & 31;
  const int wave = tid >> 5;
  const int mrow0 = blockIdx.x * 16;
  const int col0 = blockIdx.y * 256 + wave * 64;
  __builtin_prefetch(B + (size_t)col0 * KH, 0, 3);
  // stage A tile (16 x KH halves) into LDS, 16B granules
  const int vecs = 16 * KH / 8;
  const uint4* Ag = (const uint4*)(A + (size_t)mrow0 * KH);
#if USE_ASYNC_LDS
  for (int v = tid; v < vecs; v += 128) async_copy_b128(Ag + v, (uint4*)As + v);
  __builtin_amdgcn_s_wait_asynccnt(0);
#else
  for (int v = tid; v < vecs; v += 128) ((uint4*)As)[v] = Ag[v];
#endif
  __syncthreads();

  v8f acc[4] = {};
#pragma unroll 1
  for (int kt = 0; kt < KH / 32; ++kt) {
    v16h a = frag_a(As + kt * 32, KH, lane);
#pragma unroll
    for (int nt = 0; nt < 4; ++nt) {
      v16h b = frag_b(B + (size_t)(col0 + nt * 16) * KH + kt * 32, KH, lane);
      acc[nt] = __builtin_amdgcn_wmma_f32_16x16x32_f16(false, a, false, b,
                                                       (short)0, acc[nt], false, false);
    }
  }
  const int ccol = lane & 15;
  const int rofs = (lane >> 4) << 3;
#pragma unroll
  for (int nt = 0; nt < 4; ++nt) {
    int gc = col0 + nt * 16 + ccol;
    float bv = BIAS ? bias[gc] : 0.f;
#pragma unroll
    for (int i = 0; i < 8; ++i) {
      int r = mrow0 + rofs + i;
      float v = acc[nt][i] + bv;
      if (RELU) v = fmaxf(v, 0.f);
      if (OUTF16) outh[(size_t)r * ldc + gc] = (_Float16)v;
      else        outf[(size_t)r * ldc + gc] = v;
    }
  }
}

// ---------------- GRU recurrence (one workgroup, 16 waves) ----------------
// gi (4064 x 768 f32, bih already added). Per step: gh = h@Whh^T + bhh via WMMA,
// then elementwise gate math. mode 0: write h(t) to out16 slot t+1 (t<126) [update GRU
// feeding decoder input]; mode 1: write h(t) to slot t [decoder -> mem].
__global__ __launch_bounds__(512) void k_gru(const float* __restrict__ gi,
                                             const _Float16* __restrict__ Whh,
                                             const float* __restrict__ bhh,
                                             _Float16* __restrict__ out16, int mode) {
  __shared__ _Float16 h16[Sb * Dd];        // 16 KB
  __shared__ _Float16 gh[Sb * 768];        // 48 KB
  const int tid = threadIdx.x;             // 512 = 16 waves
  const int lane = tid & 31;
  const int wave = tid >> 5;
  for (int i = tid; i < Sb * Dd; i += 512) h16[i] = (_Float16)0.f;
  __syncthreads();
  const int mt = wave >> 3;                // 0..1 (rows 0-15 / 16-31)
  const int nt0 = (wave & 7) * 6;          // 6 n-tiles per wave, 48 total
  for (int t = 0; t < Nm1; ++t) {
    v8f acc[6] = {};
#pragma unroll 1
    for (int kt = 0; kt < 8; ++kt) {
      v16h a = frag_a(h16 + mt * 16 * Dd + kt * 32, Dd, lane);
#pragma unroll
      for (int q = 0; q < 6; ++q) {
        v16h b = frag_b(Whh + (size_t)((nt0 + q) * 16) * Dd + kt * 32, Dd, lane);
        acc[q] = __builtin_amdgcn_wmma_f32_16x16x32_f16(false, a, false, b,
                                                        (short)0, acc[q], false, false);
      }
    }
    const int ccol = lane & 15;
    const int rofs = (lane >> 4) << 3;
#pragma unroll
    for (int q = 0; q < 6; ++q) {
      int col = (nt0 + q) * 16 + ccol;
      float bv = bhh[col];
#pragma unroll
      for (int i = 0; i < 8; ++i) {
        int row = mt * 16 + rofs + i;
        gh[row * 768 + col] = (_Float16)(acc[q][i] + bv);
      }
    }
    __syncthreads();
    for (int idx = tid; idx < Sb * Dd; idx += 512) {
      int s = idx >> 8, d = idx & 255;
      const float* gis = gi + ((size_t)s * Nm1 + t) * 768;
      float ir = gis[d], iz = gis[256 + d], inn = gis[512 + d];
      float hr = (float)gh[s * 768 + d];
      float hz = (float)gh[s * 768 + 256 + d];
      float hn = (float)gh[s * 768 + 512 + d];
      float r = 1.f / (1.f + __expf(-(ir + hr)));
      float zz = 1.f / (1.f + __expf(-(iz + hz)));
      float n = tanhf(inn + r * hn);
      float h = (float)h16[idx];
      float hnew = (1.f - zz) * n + zz * h;
      h16[idx] = (_Float16)hnew;
      if (mode == 0) {
        if (t < Nm1 - 1) out16[((size_t)s * Nm1 + (t + 1)) * Dd + d] = (_Float16)hnew;
      } else {
        out16[((size_t)s * Nm1 + t) * Dd + d] = (_Float16)hnew;
      }
    }
    __syncthreads();
  }
}

// ---------------- fused edge readout: h0 -> WMMA layer1 -> layer2 -> BCE ----------------
// Aj = mem @ er_W0[:, :256]^T + b0 ; Bi = mem @ er_W0[:, 256:]^T (precomputed f32).
// Block: 16 edges of sample blockIdx.y, 128 threads (4 waves).
__global__ __launch_bounds__(128) void k_edge(const float* __restrict__ Aj,
                                              const float* __restrict__ Bi,
                                              const _Float16* __restrict__ W1,
                                              const float* __restrict__ b1,
                                              const float* __restrict__ W2,
                                              const float* __restrict__ b2,
                                              const float* __restrict__ con,
                                              float* __restrict__ out) {
  __shared__ _Float16 h0[16 * Dd];   // 8 KB
  __shared__ _Float16 h1[16 * 128];  // 4 KB
  __shared__ int ji[32];
  const int tid = threadIdx.x;
  const int lane = tid & 31;
  const int wave = tid >> 5;
  const int s = blockIdx.y;
  const int e0 = blockIdx.x * 16;
  if (tid < 16) {
    int eg = e0 + tid;
    int i = (int)((sqrtf(8.f * (float)eg + 1.f) - 1.f) * 0.5f);
    while ((i + 1) * (i + 2) / 2 <= eg) ++i;
    while (i * (i + 1) / 2 > eg) --i;
    ji[tid * 2 + 0] = eg - i * (i + 1) / 2;  // j
    ji[tid * 2 + 1] = i;                     // i
  }
  __syncthreads();
  for (int idx = tid; idx < 16 * Dd; idx += 128) {
    int e = idx >> 8, d = idx & 255;
    float v = Aj[((size_t)s * Nm1 + ji[e * 2]) * Dd + d] +
              Bi[((size_t)s * Nm1 + ji[e * 2 + 1]) * Dd + d];
    h0[idx] = (_Float16)fmaxf(v, 0.f);
  }
  __syncthreads();
  {  // layer1: (16x128) = relu(h0 @ W1^T + b1); 2 n-tiles per wave
    v8f acc[2] = {};
#pragma unroll 1
    for (int kt = 0; kt < 8; ++kt) {
      v16h a = frag_a(h0 + kt * 32, Dd, lane);
#pragma unroll
      for (int q = 0; q < 2; ++q) {
        v16h b = frag_b(W1 + (size_t)((wave * 2 + q) * 16) * Dd + kt * 32, Dd, lane);
        acc[q] = __builtin_amdgcn_wmma_f32_16x16x32_f16(false, a, false, b,
                                                        (short)0, acc[q], false, false);
      }
    }
    const int ccol = lane & 15;
    const int rofs = (lane >> 4) << 3;
#pragma unroll
    for (int q = 0; q < 2; ++q) {
      int col = (wave * 2 + q) * 16 + ccol;
      float bv = b1[col];
#pragma unroll
      for (int i = 0; i < 8; ++i) {
        float v = acc[q][i] + bv;
        h1[(rofs + i) * 128 + col] = (_Float16)fmaxf(v, 0.f);
      }
    }
  }
  __syncthreads();
  if (wave == 0) {  // layer2 (2 logits per edge) + softmax + BCE + reduce
    int e = lane >> 1, c = lane & 1;
    float a2 = 0.f;
    for (int k = 0; k < 128; ++k) a2 += (float)h1[e * 128 + k] * W2[c * 128 + k];
    float logit = a2 + b2[c];
    float other = __shfl_xor(logit, 1, 32);
    float m = fmaxf(logit, other);
    float es = __expf(logit - m), eo = __expf(other - m);
    float p = es / (es + eo);
    float lp = fmaxf(logf(p), -100.f);
    float l1p = fmaxf(log1pf(-p), -100.f);
    float cv = con[((size_t)s * TRIE + (e0 + e)) * 2 + c];
    float bce = -(cv * lp + (1.f - cv) * l1p);
    bce += __shfl_xor(bce, 2, 32);
    bce += __shfl_xor(bce, 4, 32);
    bce += __shfl_xor(bce, 8, 32);
    bce += __shfl_xor(bce, 16, 32);
    if (lane < 2) atomicAdd(&out[s * 2 + lane], bce);
  }
}

// ---------------- host launch ----------------

extern "C" void kernel_launch(void* const* d_in, const int* in_sizes, int n_in,
                              void* d_out, int out_size, void* d_ws, size_t ws_size,
                              hipStream_t stream) {
  (void)in_sizes; (void)n_in; (void)out_size; (void)ws_size;
  const float* x       = (const float*)d_in[0];
  const float* con     = (const float*)d_in[1];
  const float* z_ph    = (const float*)d_in[2];
  const float* init_ph = (const float*)d_in[3];
  const float* cc_W0 = (const float*)d_in[4];  const float* cc_b0 = (const float*)d_in[5];
  const float* cc_W1 = (const float*)d_in[6];  const float* cc_b1 = (const float*)d_in[7];
  const float* cc_W2 = (const float*)d_in[8];  const float* cc_b2 = (const float*)d_in[9];
  const float* er_W0 = (const float*)d_in[10]; const float* er_b0 = (const float*)d_in[11];
  const float* er_W1 = (const float*)d_in[12]; const float* er_b1 = (const float*)d_in[13];
  const float* er_W2 = (const float*)d_in[14]; const float* er_b2 = (const float*)d_in[15];
  const float* rr_W  = (const float*)d_in[16]; const float* rr_b  = (const float*)d_in[17];
  const float* dec_Wih = (const float*)d_in[18]; const float* dec_Whh = (const float*)d_in[19];
  const float* dec_bih = (const float*)d_in[20]; const float* dec_bhh = (const float*)d_in[21];
  const float* upd_Wih = (const float*)d_in[22]; const float* upd_Whh = (const float*)d_in[23];
  const float* upd_bih = (const float*)d_in[24]; const float* upd_bhh = (const float*)d_in[25];
  float* out = (float*)d_out;

  size_t off = 0;
  auto alloc = [&](size_t bytes) -> void* {
    void* p = (char*)d_ws + off;
    off += (bytes + 255) & ~(size_t)255;
    return p;
  };
  float*     c0f     = (float*)alloc(Dd * 4);
  _Float16*  uWih16  = (_Float16*)alloc((size_t)768 * Dd * 2);
  _Float16*  uWhh16  = (_Float16*)alloc((size_t)768 * Dd * 2);
  _Float16*  dWih16  = (_Float16*)alloc((size_t)768 * Dd * 2);
  _Float16*  dWhh16  = (_Float16*)alloc((size_t)768 * Dd * 2);
  _Float16*  e0j16   = (_Float16*)alloc((size_t)Dd * Dd * 2);
  _Float16*  e0i16   = (_Float16*)alloc((size_t)Dd * Dd * 2);
  _Float16*  e1_16   = (_Float16*)alloc((size_t)128 * Dd * 2);
  _Float16*  rrW16   = (_Float16*)alloc((size_t)Dd * FPAD * 2);
  _Float16*  xp16    = (_Float16*)alloc((size_t)Mrows * FPAD * 2);
  _Float16*  rn16    = (_Float16*)alloc((size_t)Mrows * Dd * 2);
  float*     gi_u    = (float*)alloc((size_t)Mrows * 768 * 4);
  _Float16*  cont16  = (_Float16*)alloc((size_t)Mrows * Dd * 2);
  float*     gi_d    = (float*)alloc((size_t)Mrows * 768 * 4);
  _Float16*  mem16   = (_Float16*)alloc((size_t)Mrows * Dd * 2);
  float*     Aj      = (float*)alloc((size_t)Mrows * Dd * 4);
  float*     Bi      = (float*)alloc((size_t)Mrows * Dd * 4);

  k_zero<<<1, 64, 0, stream>>>(out);
  k_c0<<<1, 256, 0, stream>>>(z_ph, init_ph, cc_W0, cc_b0, cc_W1, cc_b1,
                              cc_W2, cc_b2, c0f, cont16);
  // weight conversions to f16 workspace
  k_cvt16<<<768, 256, 0, stream>>>(upd_Wih, 256, uWih16, 256, 768, 256);
  k_cvt16<<<768, 256, 0, stream>>>(upd_Whh, 256, uWhh16, 256, 768, 256);
  k_cvt16<<<768, 256, 0, stream>>>(dec_Wih, 256, dWih16, 256, 768, 256);
  k_cvt16<<<768, 256, 0, stream>>>(dec_Whh, 256, dWhh16, 256, 768, 256);
  k_cvt16<<<256, 256, 0, stream>>>(er_W0, 512, e0j16, 256, 256, 256);
  k_cvt16<<<256, 256, 0, stream>>>(er_W0 + 256, 512, e0i16, 256, 256, 256);
  k_cvt16<<<128, 256, 0, stream>>>(er_W1, 256, e1_16, 256, 128, 256);
  k_cvt16<<<544, 256, 0, stream>>>(rr_W, 516, rrW16, 544, 256, 516);
  k_cvt_x<<<4096, 256, 0, stream>>>(x, xp16);

  // rnninp = relu(x[:,1:] @ rr_W^T + rr_b)  (f16 out)
  k_gemm16<FPAD, true, true, true><<<dim3(Mrows / 16, 1), 128, 0, stream>>>(
      xp16, rrW16, rr_b, rn16, nullptr, Dd);
  // update-GRU input gates: gi_u = rnninp @ upd_Wih^T + upd_bih
  k_gemm16<Dd, false, true, false><<<dim3(Mrows / 16, 3), 128, 0, stream>>>(
      rn16, uWih16, upd_bih, nullptr, gi_u, 768);
  // update GRU -> fills cont16 slots 1..126 (slot 0 = c0 from k_c0)
  k_gru<<<1, 512, 0, stream>>>(gi_u, uWhh16, upd_bhh, cont16, 0);
  // decoder input gates: gi_d = content_seq @ dec_Wih^T + dec_bih
  k_gemm16<Dd, false, true, false><<<dim3(Mrows / 16, 3), 128, 0, stream>>>(
      cont16, dWih16, dec_bih, nullptr, gi_d, 768);
  // decoder GRU -> mem16
  k_gru<<<1, 512, 0, stream>>>(gi_d, dWhh16, dec_bhh, mem16, 1);
  // edge layer-0 factorization: Aj = mem@W0j^T + b0 ; Bi = mem@W0i^T
  k_gemm16<Dd, false, true, false><<<dim3(Mrows / 16, 1), 128, 0, stream>>>(
      mem16, e0j16, er_b0, nullptr, Aj, Dd);
  k_gemm16<Dd, false, false, false><<<dim3(Mrows / 16, 1), 128, 0, stream>>>(
      mem16, e0i16, nullptr, nullptr, Bi, Dd);
  // fused edge readout + BCE accumulation
  k_edge<<<dim3(TRIE / 16, Sb), 128, 0, stream>>>(Aj, Bi, e1_16, er_b1,
                                                  er_W2, er_b2, con, out);
}
